// QuantizedLinear_17695265260025
// MI455X (gfx1250) — matrix operand order repaired and checked by reading, not compiled, not run on hardware
//
#include <hip/hip_runtime.h>
#include <stdint.h>

typedef __attribute__((ext_vector_type(4)))  _Float16 v4h;
typedef __attribute__((ext_vector_type(8)))  _Float16 v8h;
typedef __attribute__((ext_vector_type(16))) _Float16 v16h;
typedef __attribute__((ext_vector_type(8)))  float    v8f;

#define IN_F   4096
#define OUT_F  14336
#define BM     128
#define BN     256
#define BK     32
#define KSTEPS (IN_F / BK)          // 128
#define XSTR   (BK + 8)             // 40 halves = 80B row stride (16B aligned, bank-spread)
#define QSTR   (BK + 8)

// Decode 8 packed int4 (one u32: byte j -> K=2j (hi nibble), K=2j+1 (lo nibble))
// into exact f16 integers via exponent-bias trick: as_f16((n^8)|0x6400) == 1024+(v+8)
__device__ __forceinline__ void decode8(uint32_t w, _Float16* r) {
#pragma unroll
  for (int j = 0; j < 4; ++j) {
    uint32_t b  = (w >> (8 * j)) & 0xffu;
    uint16_t hi = (uint16_t)(((b >> 4) ^ 8u) | 0x6400u);
    uint16_t lo = (uint16_t)(((b & 15u) ^ 8u) | 0x6400u);
    r[2 * j]     = __builtin_bit_cast(_Float16, hi) - (_Float16)1032.0f;
    r[2 * j + 1] = __builtin_bit_cast(_Float16, lo) - (_Float16)1032.0f;
  }
}

__global__ __launch_bounds__(256) void qlin4_wmma(
    const float* __restrict__ x, const uint8_t* __restrict__ qw,
    const float* __restrict__ scale, const float* __restrict__ bias,
    float* __restrict__ out) {
  __shared__ _Float16 sX[2][BM][XSTR];   // 20 KB : x tile, f16
  __shared__ _Float16 sQ[2][BN][QSTR];   // 40 KB : dequantized weights, f16

  const int tid  = threadIdx.x;
  const int lane = tid & 31;
  const int lh   = lane & 15;
  const int hiL  = lane >> 4;            // 0 or 1
  const int wave = tid >> 5;
  const int waveM = wave & 1;            // 2 waves along M (64 rows each)
  const int waveN = wave >> 1;           // 4 waves along N (64 cols each)

  const int m0 = blockIdx.y * BM;
  const int n0 = blockIdx.x * BN;

  v8f acc[4][4] = {};                    // 64x64 wave tile -> 4x4 16x16 subtiles

  // -------- global prefetch registers --------
  float4 xr[4];                          // 128x32 f32 tile: 1024 float4, 4/thread
  uint4  qr;                             // 256 rows x 16B packed: 16B/thread

  const float*   xg = x  + (size_t)(m0 + (tid >> 3)) * IN_F + (tid & 7) * 4;
  const uint8_t* qg = qw + (size_t)(n0 + tid) * (IN_F / 2);

  auto load_global = [&](int kt) {
    const int k0 = kt * BK;
#pragma unroll
    for (int i = 0; i < 4; ++i)          // rows tid>>3, +32, +64, +96
      xr[i] = *(const float4*)(xg + (size_t)i * 32 * IN_F + k0);
    qr = *(const uint4*)(qg + (k0 >> 1));
  };

  auto store_lds = [&](int buf) {
#pragma unroll
    for (int i = 0; i < 4; ++i) {
      int row = (tid >> 3) + i * 32;
      int c4  = tid & 7;
      v4h h;
      h[0] = (_Float16)xr[i].x; h[1] = (_Float16)xr[i].y;
      h[2] = (_Float16)xr[i].z; h[3] = (_Float16)xr[i].w;
      *(v4h*)&sX[buf][row][c4 * 4] = h;
    }
    {
      union { v8h v[4]; _Float16 h[32]; } u;
      decode8(qr.x, &u.h[0]);
      decode8(qr.y, &u.h[8]);
      decode8(qr.z, &u.h[16]);
      decode8(qr.w, &u.h[24]);
      _Float16* Qd = &sQ[buf][tid][0];
      *(v8h*)(Qd)      = u.v[0];
      *(v8h*)(Qd + 8)  = u.v[1];
      *(v8h*)(Qd + 16) = u.v[2];
      *(v8h*)(Qd + 24) = u.v[3];
    }
  };

  load_global(0);
  store_lds(0);
  __syncthreads();

  for (int kt = 0; kt < KSTEPS; ++kt) {
    const int  cur  = kt & 1;
    const bool more = (kt + 1) < KSTEPS;
    if (more) load_global(kt + 1);

    // deep prefetch into L2 (global_prefetch_b8): +4 k-tiles ahead
    if (kt + 4 < KSTEPS) {
      __builtin_prefetch(xg + (size_t)(kt + 4) * BK, 0, 1);
      __builtin_prefetch(qg + (size_t)(kt + 4) * (BK / 2), 0, 1);
    }

    // ---- A fragments (16x32 f16): lanes0-15 K{0..7,16..23}, lanes16-31 K{8..15,24..31}
    v16h a[4];
#pragma unroll
    for (int s = 0; s < 4; ++s) {
      const _Float16* Xr = &sX[cur][waveM * 64 + s * 16 + lh][0];
      const int kb = hiL * 8;
      union { v16h v; v8h h[2]; } u;
      u.h[0] = *(const v8h*)(Xr + kb);
      u.h[1] = *(const v8h*)(Xr + kb + 16);
      a[s] = u.v;
    }

    // ---- B fragments (32x16 f16): lanes0-15 K{0..7,16..23}, lanes16-31 K{8..15,24..31} ----
    // Build all four up front: one dscnt wait, then 16 back-to-back WMMAs.
    v16h b[4];
#pragma unroll
    for (int t = 0; t < 4; ++t) {
      const _Float16* Qr = &sQ[cur][waveN * 64 + t * 16 + lh][0];
      const int kb = hiL * 8;
      union { v16h v; v8h h[2]; } u;
      u.h[0] = *(const v8h*)(Qr + kb);
      u.h[1] = *(const v8h*)(Qr + kb + 16);
      b[t] = u.v;
    }

    // ---- 16 WMMAs: D = A x B + C ----
#pragma unroll
    for (int t = 0; t < 4; ++t)
#pragma unroll
      for (int s = 0; s < 4; ++s)
        acc[s][t] = __builtin_amdgcn_wmma_f32_16x16x32_f16(
            false, a[s], false, b[t], (short)0, acc[s][t], false, false);

    if (more) {
      store_lds((kt + 1) & 1);
      __syncthreads();
    }
  }

  // -------- epilogue: out = acc*scale[n] + bias[n] --------
#pragma unroll
  for (int t = 0; t < 4; ++t) {
    const int n  = n0 + waveN * 64 + t * 16 + lh;
    const float sc = scale[n];
    const float bi = bias[n];
#pragma unroll
    for (int s = 0; s < 4; ++s) {
      const int mbase = m0 + waveM * 64 + s * 16 + hiL * 8;
      float* op = out + (size_t)mbase * OUT_F + n;
#pragma unroll
      for (int r = 0; r < 8; ++r)
        op[(size_t)r * OUT_F] = acc[s][t][r] * sc + bi;
    }
  }
}

extern "C" void kernel_launch(void* const* d_in, const int* in_sizes, int n_in,
                              void* d_out, int out_size, void* d_ws, size_t ws_size,
                              hipStream_t stream) {
  const float*   xp = (const float*)d_in[0];
  const uint8_t* qp = (const uint8_t*)d_in[1];
  const float*   sp = (const float*)d_in[2];
  const float*   bp = (const float*)d_in[3];
  float*         op = (float*)d_out;

  const int M = in_sizes[0] / IN_F;          // 8192
  dim3 grid(OUT_F / BN, M / BM);             // 56 x 64
  qlin4_wmma<<<grid, 256, 0, stream>>>(xp, qp, sp, bp, op);
}
